// LlamaFlashAttention_MUSTAFAR_75952201663290
// MI455X (gfx1250) — matrix operand order repaired and checked by reading, not compiled, not run on hardware
//
#include <hip/hip_runtime.h>

// ---------------------------------------------------------------------------
// Types for CDNA5 WMMA (gfx1250, wave32): D(16x16 f32) = A(16x32 bf16) x B(32x16 bf16) + C
// ---------------------------------------------------------------------------
typedef __bf16 bf16;
typedef __attribute__((ext_vector_type(16))) __bf16 v16bf;
typedef __attribute__((ext_vector_type(8)))  __bf16 v8bf;
typedef __attribute__((ext_vector_type(8)))  float  v8f;

#define WMMA_BF16(A, B, C) \
  __builtin_amdgcn_wmma_f32_16x16x32_bf16(false, (A), false, (B), (short)0, (C), false, false)

static __device__ __forceinline__ v8f vzero8() {
  v8f z;
#pragma unroll
  for (int i = 0; i < 8; ++i) z[i] = 0.0f;
  return z;
}

static __device__ __forceinline__ v16bf pack16(v8bf lo, v8bf hi) {
  v16bf r;
#pragma unroll
  for (int i = 0; i < 8; ++i) { r[i] = lo[i]; r[i + 8] = hi[i]; }
  return r;
}

static __device__ __forceinline__ v16bf load_frag(const bf16* p0, const bf16* p1) {
  return pack16(*(const v8bf*)p0, *(const v8bf*)p1);
}

// ---- CDNA5 async global->LDS (tracked by ASYNCcnt, in-order completion) ----
static __device__ __forceinline__ void async_ld_b128(unsigned lds_off, const void* g) {
  asm volatile("global_load_async_to_lds_b128 %0, %1, off" :: "v"(lds_off), "v"(g) : "memory");
}
#define S_WAIT_ASYNCCNT(n) asm volatile("s_wait_asynccnt " #n ::: "memory")

// Low 32 bits of a flat shared pointer are the LDS byte offset (ISA aperture rule).
static __device__ __forceinline__ unsigned lds_off_of(const void* p) {
  return (unsigned)(unsigned long long)p;
}

// ---------------------------------------------------------------------------
// Model constants (reference: B=1, T=2048)
// ---------------------------------------------------------------------------
#define TT      2048
#define HID     4096
#define NH      32
#define NKV     8
#define HD      128
#define KVDIM   (NKV * HD)     // 1024
#define KEEP_K  89             // max(1, int(0.7*128)) = 89
#define LN_THETA 9.210340371976184f  // ln(10000)

// ---------------------------------------------------------------------------
// fp32 -> bf16 conversion (grid-stride)
// ---------------------------------------------------------------------------
__global__ void cvt_f32_bf16_kernel(const float* __restrict__ s, bf16* __restrict__ d, size_t n) {
  size_t i = (size_t)blockIdx.x * blockDim.x + threadIdx.x;
  size_t stride = (size_t)gridDim.x * blockDim.x;
  for (; i < n; i += stride) d[i] = (bf16)s[i];
}

// ---------------------------------------------------------------------------
// C[M,N] (f32) = A[M,K](bf16 row major) * W[N,K]^T (bf16 row major)
// 256 threads = 8 waves; each wave owns a 32x64 tile: 2 A-frags x 4 B-frags
// = 8 independent WMMA accumulator chains. Rotated 2x-unrolled software
// pipeline: each WMMA half-block consumes fragments loaded a half-iteration
// earlier (no register copies, no bottom-of-loop drain); addresses advance by
// pointer bumps on dedicated registers (no WAR hazards with WMMA sources).
// Requires K % 64 == 0. Block tile 64x256; grid (M/64, N/256).
// NOTE: final half-iteration prefetch over-reads <=96B past A — callers pad.
// ---------------------------------------------------------------------------
__global__ void gemm_bf16_wmma_kernel(const bf16* __restrict__ A, const bf16* __restrict__ W,
                                      float* __restrict__ C, int M, int N, int K) {
  const int lane = threadIdx.x & 31;
  const int wave = threadIdx.x >> 5;
  const int half = lane >> 4;
  const int l15  = lane & 15;
  const int m0 = blockIdx.x * 64 + (wave >> 2) * 32;
  const int n0 = blockIdx.y * 256 + (wave & 3) * 64;

  v8f acc[8];
#pragma unroll
  for (int j = 0; j < 8; ++j) acc[j] = vzero8();

  const bf16* pa0 = A + (size_t)(m0 + l15) * K + half * 16;
  const bf16* pa1 = pa0 + (size_t)16 * K;
  const bf16* pw0 = W + (size_t)(n0 + 0 * 16 + l15) * K + half * 16;
  const bf16* pw1 = W + (size_t)(n0 + 1 * 16 + l15) * K + half * 16;
  const bf16* pw2 = W + (size_t)(n0 + 2 * 16 + l15) * K + half * 16;
  const bf16* pw3 = W + (size_t)(n0 + 3 * 16 + l15) * K + half * 16;

  // phase-A fragments (k = 0)
  v16bf fa0 = load_frag(pa0, pa0 + 8);
  v16bf fa1 = load_frag(pa1, pa1 + 8);
  v16bf fb0 = load_frag(pw0, pw0 + 8);
  v16bf fb1 = load_frag(pw1, pw1 + 8);
  v16bf fb2 = load_frag(pw2, pw2 + 8);
  v16bf fb3 = load_frag(pw3, pw3 + 8);

  for (int it = 0, nit = K >> 6; it < nit; ++it) {
    // phase-B fragments (k + 32)
    v16bf ga0 = load_frag(pa0 + 32, pa0 + 40);
    v16bf ga1 = load_frag(pa1 + 32, pa1 + 40);
    v16bf gb0 = load_frag(pw0 + 32, pw0 + 40);
    v16bf gb1 = load_frag(pw1 + 32, pw1 + 40);
    v16bf gb2 = load_frag(pw2 + 32, pw2 + 40);
    v16bf gb3 = load_frag(pw3 + 32, pw3 + 40);
    __builtin_prefetch(pa0 + 192, 0, 1);

    acc[0] = WMMA_BF16(fa0, fb0, acc[0]);
    acc[4] = WMMA_BF16(fa1, fb0, acc[4]);
    acc[1] = WMMA_BF16(fa0, fb1, acc[1]);
    acc[5] = WMMA_BF16(fa1, fb1, acc[5]);
    acc[2] = WMMA_BF16(fa0, fb2, acc[2]);
    acc[6] = WMMA_BF16(fa1, fb2, acc[6]);
    acc[3] = WMMA_BF16(fa0, fb3, acc[3]);
    acc[7] = WMMA_BF16(fa1, fb3, acc[7]);

    pa0 += 64; pa1 += 64;
    pw0 += 64; pw1 += 64; pw2 += 64; pw3 += 64;

    // phase-A fragments for the next iteration (new base; last iter over-reads)
    fa0 = load_frag(pa0, pa0 + 8);
    fa1 = load_frag(pa1, pa1 + 8);
    fb0 = load_frag(pw0, pw0 + 8);
    fb1 = load_frag(pw1, pw1 + 8);
    fb2 = load_frag(pw2, pw2 + 8);
    fb3 = load_frag(pw3, pw3 + 8);

    acc[0] = WMMA_BF16(ga0, gb0, acc[0]);
    acc[4] = WMMA_BF16(ga1, gb0, acc[4]);
    acc[1] = WMMA_BF16(ga0, gb1, acc[1]);
    acc[5] = WMMA_BF16(ga1, gb1, acc[5]);
    acc[2] = WMMA_BF16(ga0, gb2, acc[2]);
    acc[6] = WMMA_BF16(ga1, gb2, acc[6]);
    acc[3] = WMMA_BF16(ga0, gb3, acc[3]);
    acc[7] = WMMA_BF16(ga1, gb3, acc[7]);
  }

#pragma unroll
  for (int j = 0; j < 4; ++j)
#pragma unroll
    for (int r = 0; r < 8; ++r) {
      const int col = n0 + j * 16 + l15;
      C[(size_t)(m0 + r + 8 * half) * N + col]      = acc[j][r];
      C[(size_t)(m0 + 16 + r + 8 * half) * N + col] = acc[4 + j][r];
    }
}

// ---------------------------------------------------------------------------
// RoPE on Q: qf[t][h*128+d] (f32) -> qh[h][t][d] (bf16)
// ---------------------------------------------------------------------------
__global__ void rope_q_kernel(const float* __restrict__ qf, const int* __restrict__ pos,
                              bf16* __restrict__ qh) {
  size_t total = (size_t)TT * NH * HD;
  size_t stride = (size_t)gridDim.x * blockDim.x;
  for (size_t idx = (size_t)blockIdx.x * blockDim.x + threadIdx.x; idx < total; idx += stride) {
    int d = (int)(idx & (HD - 1));
    int h = (int)((idx >> 7) & (NH - 1));
    int t = (int)(idx >> 12);
    const float* row = qf + (size_t)t * HID + h * HD;
    float x  = row[d];
    int  dp  = (d < 64) ? d + 64 : d - 64;
    float rot = (d < 64) ? -row[dp] : row[dp];
    float inv = __expf(-(float)(2 * (d & 63)) * (LN_THETA / (float)HD));
    float ang = (float)pos[t] * inv;
    float sn, cs;
    __sincosf(ang, &sn, &cs);
    qh[((size_t)h * TT + t) * HD + d] = (bf16)(x * cs + rot * sn);
  }
}

// ---------------------------------------------------------------------------
// RoPE + magnitude prune on K: kf[t][kv*128+d] -> kh[kv][t][d] (bf16).
// keep(x_i) <=> #{j : |x_j| <= |x_i|} >= KEEP_K  (exact kth-value rule)
// ---------------------------------------------------------------------------
__global__ void rope_prune_k_kernel(const float* __restrict__ kf, const int* __restrict__ pos,
                                    bf16* __restrict__ kh) {
  __shared__ float sabs[4][HD];
  const int wave = threadIdx.x >> 5;
  const int lane = threadIdx.x & 31;
  const int row  = blockIdx.x * 4 + wave;
  const int t  = row >> 3;
  const int kv = row & 7;
  const float* src = kf + (size_t)t * KVDIM + kv * HD;

  float r[4], a[4];
#pragma unroll
  for (int j = 0; j < 4; ++j) {
    int d = lane * 4 + j;
    float x  = src[d];
    int  dp  = (d < 64) ? d + 64 : d - 64;
    float rot = (d < 64) ? -src[dp] : src[dp];
    float inv = __expf(-(float)(2 * (d & 63)) * (LN_THETA / (float)HD));
    float ang = (float)pos[t] * inv;
    float sn, cs;
    __sincosf(ang, &sn, &cs);
    r[j] = x * cs + rot * sn;
    a[j] = fabsf(r[j]);
    sabs[wave][d] = a[j];
  }
  __syncthreads();

  int cnt[4] = {0, 0, 0, 0};
  for (int q = 0; q < HD; ++q) {
    float aq = sabs[wave][q];
#pragma unroll
    for (int j = 0; j < 4; ++j) cnt[j] += (aq <= a[j]) ? 1 : 0;
  }

  bf16* dst = kh + ((size_t)kv * TT + t) * HD + lane * 4;
#pragma unroll
  for (int j = 0; j < 4; ++j) dst[j] = (cnt[j] >= KEEP_K) ? (bf16)r[j] : (bf16)0.0f;
}

// ---------------------------------------------------------------------------
// Magnitude prune on V, writing V transposed: vf[t][kv*128+d] -> vt[kv][d][t]
// ---------------------------------------------------------------------------
__global__ void prune_v_kernel(const float* __restrict__ vf, bf16* __restrict__ vt) {
  __shared__ float sabs[4][HD];
  const int wave = threadIdx.x >> 5;
  const int lane = threadIdx.x & 31;
  const int row  = blockIdx.x * 4 + wave;
  const int t  = row >> 3;
  const int kv = row & 7;
  const float* src = vf + (size_t)t * KVDIM + kv * HD;

  float x[4], a[4];
#pragma unroll
  for (int j = 0; j < 4; ++j) {
    int d = lane * 4 + j;
    x[j] = src[d];
    a[j] = fabsf(x[j]);
    sabs[wave][d] = a[j];
  }
  __syncthreads();

  int cnt[4] = {0, 0, 0, 0};
  for (int q = 0; q < HD; ++q) {
    float aq = sabs[wave][q];
#pragma unroll
    for (int j = 0; j < 4; ++j) cnt[j] += (aq <= a[j]) ? 1 : 0;
  }
#pragma unroll
  for (int j = 0; j < 4; ++j) {
    int d = lane * 4 + j;
    vt[((size_t)kv * HD + d) * TT + t] = (cnt[j] >= KEEP_K) ? (bf16)x[j] : (bf16)0.0f;
  }
}

// ---------------------------------------------------------------------------
// Flash attention with WMMA + async global->LDS K/V staging.
// Grid (T/32, NH), block 64 = 2 waves. Each wave owns a 16-row Q tile and
// streams 32-key tiles. K double-buffered in LDS, V single-buffered;
// ASYNCcnt in-order completion enables partial waits:
//   issue V[kt], issue K[kt+1]  -> s_wait_asynccnt 32  => K[kt] resident
//   ... scores + softmax ...    -> s_wait_asynccnt 16  => V[kt] resident
// qh:[h][t][d]  kh:[kv][t][d]  vt:[kv][d][t]  out attnb:[t][h*128+d] (bf16)
// ---------------------------------------------------------------------------
#define AW 2
__global__ void attn_wmma_kernel(const bf16* __restrict__ qh, const bf16* __restrict__ kh,
                                 const bf16* __restrict__ vt, bf16* __restrict__ attnb) {
  __shared__ bf16 kbuf[AW][2][32 * HD];   // 2 x 8KB per wave
  __shared__ bf16 vbuf[AW][HD * 32];      // 8KB per wave ([d][key] row-major)
  __shared__ bf16 plds[AW][16 * 32];      // P-tile staging for C->A transpose

  const int lane = threadIdx.x & 31;
  const int wave = threadIdx.x >> 5;
  const int half = lane >> 4;
  const int l15  = lane & 15;
  const int h    = blockIdx.y;
  const int kv   = h >> 2;                // GQA: 4 query heads per KV head
  const int qt0  = blockIdx.x * (16 * AW) + wave * 16;

  // Q tile 16x128 -> 4 A-fragments, kept in registers for the whole kernel.
  v16bf qa[4];
  {
    const bf16* qrow = qh + ((size_t)h * TT + (qt0 + l15)) * HD;
#pragma unroll
    for (int c = 0; c < 4; ++c)
      qa[c] = load_frag(qrow + c * 32 + half * 16, qrow + c * 32 + half * 16 + 8);
  }

  v8f o[8];
#pragma unroll
  for (int dc = 0; dc < 8; ++dc) o[dc] = vzero8();
  float m[8], l[8];
#pragma unroll
  for (int r = 0; r < 8; ++r) { m[r] = -1e30f; l[r] = 0.0f; }

  const float scale = 0.08838834764831845f;  // 1/sqrt(128)
  const int ntiles = (qt0 + 15) / 32 + 1;    // causal: keys <= qt0+15
  const bf16* kvbase = kh + (size_t)kv * TT * HD;   // K tile kt: 8KB contiguous
  const bf16* vtbase = vt + (size_t)kv * HD * TT;   // V row d: 64B @ stride 4KB

  // ---- prologue: async-stage K[0] into LDS buffer 0 (16 x b128 per lane)
  {
    const unsigned kl = lds_off_of(&kbuf[wave][0][0]);
    const char* gk = (const char*)kvbase;
#pragma unroll
    for (int i = 0; i < 16; ++i) {
      const unsigned boff = (unsigned)((i * 32 + lane) * 16);
      async_ld_b128(kl + boff, gk + boff);
    }
  }

  for (int kt = 0; kt < ntiles; ++kt) {
    const int k0 = kt * 32;
    const int cb = kt & 1;

    // ---- issue async V[kt] ([d=0..127][key k0..k0+31], 16B chunks)
    {
      const unsigned vl = lds_off_of(&vbuf[wave][0]);
#pragma unroll
      for (int i = 0; i < 16; ++i) {
        const int c   = i * 32 + lane;          // 0..511
        const int row = c >> 2;                 // d
        const int sub = (c & 3) * 16;           // byte offset within 64B row
        const char* g = (const char*)(vtbase + (size_t)row * TT + k0) + sub;
        async_ld_b128(vl + (unsigned)(c * 16), g);
      }
    }
    // ---- issue async K[kt+1] into the other buffer (unconditional; the
    //      <=8KB over-read past kh lands in the next workspace buffer)
    {
      const unsigned kl = lds_off_of(&kbuf[wave][cb ^ 1][0]);
      const char* gk = (const char*)(kvbase + (size_t)(k0 + 32) * HD);
#pragma unroll
      for (int i = 0; i < 16; ++i) {
        const unsigned boff = (unsigned)((i * 32 + lane) * 16);
        async_ld_b128(kl + boff, gk + boff);
      }
    }

    // ---- K[kt] resident (oldest 16 of <=48 outstanding; in-order completion)
    S_WAIT_ASYNCCNT(32);

    // ---- scores: S[16x32] = Q x K^T from LDS (ds_load_b128 fragments)
    v8f s0 = vzero8(), s1 = vzero8();
#pragma unroll
    for (int c = 0; c < 4; ++c) {
      const bf16* kp0 = &kbuf[wave][cb][l15 * HD + c * 32 + half * 16];
      s0 = WMMA_BF16(qa[c], load_frag(kp0, kp0 + 8), s0);
      const bf16* kp1 = kp0 + 16 * HD;
      s1 = WMMA_BF16(qa[c], load_frag(kp1, kp1 + 8), s1);
    }

    // ---- causal mask + online softmax (16-lane row reductions)
#pragma unroll
    for (int r = 0; r < 8; ++r) {
      const int qrow = qt0 + r + 8 * half;
      float x0 = s0[r] * scale; if (k0 + l15 > qrow)      x0 = -1e9f;
      float x1 = s1[r] * scale; if (k0 + 16 + l15 > qrow) x1 = -1e9f;

      float mx = fmaxf(x0, x1);
#pragma unroll
      for (int off = 1; off < 16; off <<= 1) mx = fmaxf(mx, __shfl_xor(mx, off, 32));
      const float nm = fmaxf(m[r], mx);
      const float f  = __expf(m[r] - nm);
      const float p0 = __expf(x0 - nm);
      const float p1 = __expf(x1 - nm);
      float rs = p0 + p1;
#pragma unroll
      for (int off = 1; off < 16; off <<= 1) rs += __shfl_xor(rs, off, 32);
      l[r] = l[r] * f + rs;
      m[r] = nm;
#pragma unroll
      for (int dc = 0; dc < 8; ++dc) o[dc][r] *= f;

      const int rrow = r + 8 * half;
      plds[wave][rrow * 32 + l15]      = (bf16)p0;
      plds[wave][rrow * 32 + 16 + l15] = (bf16)p1;
    }

    // ---- P (C layout) -> A-fragment via LDS (DS ops in-order per wave)
    const bf16* pr = &plds[wave][l15 * 32 + half * 16];
    v16bf pa = pack16(*(const v8bf*)pr, *(const v8bf*)(pr + 8));

    // ---- V[kt] resident (K[kt+1] may still be in flight)
    S_WAIT_ASYNCCNT(16);

    // ---- O += P x V from LDS
#pragma unroll
    for (int dc = 0; dc < 8; ++dc) {
      const bf16* vp = &vbuf[wave][(dc * 16 + l15) * 32 + half * 16];
      o[dc] = WMMA_BF16(pa, load_frag(vp, vp + 8), o[dc]);
    }
  }

  // ---- epilogue: normalize and store [t][h*128+d] bf16
#pragma unroll
  for (int dc = 0; dc < 8; ++dc)
#pragma unroll
    for (int r = 0; r < 8; ++r) {
      const int t = qt0 + r + 8 * half;
      attnb[(size_t)t * HID + h * HD + dc * 16 + l15] = (bf16)(o[dc][r] / l[r]);
    }
}

// ---------------------------------------------------------------------------
// Host-side orchestration
// ---------------------------------------------------------------------------
extern "C" void kernel_launch(void* const* d_in, const int* in_sizes, int n_in,
                              void* d_out, int out_size, void* d_ws, size_t ws_size,
                              hipStream_t stream) {
  (void)in_sizes; (void)n_in; (void)out_size; (void)ws_size;
  const float* hs  = (const float*)d_in[0];
  const int*   pos = (const int*)d_in[1];
  const float* wq  = (const float*)d_in[2];
  const float* wk  = (const float*)d_in[3];
  const float* wv  = (const float*)d_in[4];
  const float* wo  = (const float*)d_in[5];
  float* out = (float*)d_out;

  char* ws = (char*)d_ws;
  size_t off = 0;
  auto alloc = [&](size_t bytes) -> char* {
    char* p = ws + off;
    off += (bytes + 255) & ~(size_t)255;
    return p;
  };

  bf16* hb    = (bf16*)alloc((size_t)TT * HID * 2);
  bf16* wqb   = (bf16*)alloc((size_t)HID * HID * 2);
  bf16* wkb   = (bf16*)alloc((size_t)KVDIM * HID * 2);
  bf16* wvb   = (bf16*)alloc((size_t)KVDIM * HID * 2);
  bf16* wob   = (bf16*)alloc((size_t)HID * HID * 2);
  float* qf   = (float*)alloc((size_t)TT * HID * 4);
  float* kf   = (float*)alloc((size_t)TT * KVDIM * 4);
  float* vf   = (float*)alloc((size_t)TT * KVDIM * 4);
  bf16* qhb   = (bf16*)alloc((size_t)NH * TT * HD * 2);
  bf16* khb   = (bf16*)alloc((size_t)NKV * TT * HD * 2);
  bf16* vtb   = (bf16*)alloc((size_t)NKV * HD * TT * 2);   // follows khb: absorbs over-read
  bf16* attnb = (bf16*)alloc((size_t)TT * HID * 2);
  (void)alloc(4096);  // pad: absorbs gemm rotated-pipeline over-read past attnb

  // 1) fp32 -> bf16
  cvt_f32_bf16_kernel<<<4096, 256, 0, stream>>>(hs, hb, (size_t)TT * HID);
  cvt_f32_bf16_kernel<<<4096, 256, 0, stream>>>(wq, wqb, (size_t)HID * HID);
  cvt_f32_bf16_kernel<<<4096, 256, 0, stream>>>(wk, wkb, (size_t)KVDIM * HID);
  cvt_f32_bf16_kernel<<<4096, 256, 0, stream>>>(wv, wvb, (size_t)KVDIM * HID);
  cvt_f32_bf16_kernel<<<4096, 256, 0, stream>>>(wo, wob, (size_t)HID * HID);

  // 2) QKV projections (bf16 WMMA, f32 out); block tile 64x256
  gemm_bf16_wmma_kernel<<<dim3(TT / 64, HID / 256), 256, 0, stream>>>(hb, wqb, qf, TT, HID, HID);
  gemm_bf16_wmma_kernel<<<dim3(TT / 64, KVDIM / 256), 256, 0, stream>>>(hb, wkb, kf, TT, KVDIM, HID);
  gemm_bf16_wmma_kernel<<<dim3(TT / 64, KVDIM / 256), 256, 0, stream>>>(hb, wvb, vf, TT, KVDIM, HID);

  // 3) RoPE + Mustafar pruning
  rope_q_kernel<<<8192, 256, 0, stream>>>(qf, pos, qhb);
  rope_prune_k_kernel<<<TT * NKV / 4, 128, 0, stream>>>(kf, pos, khb);
  prune_v_kernel<<<TT * NKV / 4, 128, 0, stream>>>(vf, vtb);

  // 4) causal GQA flash attention (WMMA + async LDS staging)
  attn_wmma_kernel<<<dim3(TT / (16 * AW), NH), 32 * AW, 0, stream>>>(qhb, khb, vtb, attnb);

  // 5) output projection -> f32 d_out
  gemm_bf16_wmma_kernel<<<dim3(TT / 64, HID / 256), 256, 0, stream>>>(attnb, wob, out, TT, HID, HID);
}
